// Attention_24343874634217
// MI455X (gfx1250) — matrix operand order repaired and checked by reading, not compiled, not run on hardware
//
#include <hip/hip_runtime.h>

#define DIM   1024
#define NH    16
#define HD    64
#define BATCH 4
#define SEQ   4096
#define BT    (BATCH * SEQ)
#define EPSF  1.1920928955078125e-07f

typedef __attribute__((ext_vector_type(16))) __bf16 v16bf;
typedef __attribute__((ext_vector_type(8)))  __bf16 v8bf;
typedef __attribute__((ext_vector_type(8)))  float  v8f;

__device__ __forceinline__ v8f wmma_bf16(v16bf a, v16bf b, v8f c) {
  // D = A(16x32 bf16) * B(32x16 bf16) + C(16x16 f32)
  return __builtin_amdgcn_wmma_f32_16x16x32_bf16(false, a, false, b,
                                                 (short)0, c, false, false);
}

// A-fragment K mapping (16-bit A 16x32, ISA 7.12.2):
// element j of lane L holds K = (j>>3)*16 + (L>>4)*8 + (j&7), row M = L&15.
__device__ __forceinline__ int a_k(int j, int half) {
  return ((j >> 3) << 4) + (half << 3) + (j & 7);
}

// CDNA5 async copy global->LDS (ISA 15.18.3 op 98, ASYNCcnt-tracked).
// LDS destination = low 32 bits of the generic pointer (aperture rule, 10.2).
__device__ __forceinline__ void async_load_b128(void* lds, const void* gptr) {
  unsigned loff = (unsigned)(unsigned long long)lds;
  unsigned long long ga = (unsigned long long)gptr;
  asm volatile("global_load_async_to_lds_b128 %0, %1, off"
               :: "v"(loff), "v"(ga) : "memory");
}
__device__ __forceinline__ void wait_async0() {
  asm volatile("s_wait_asynccnt 0x0" ::: "memory");
}

// ---------------------------------------------------------------------------
// Kernel 1: q/k/v = x @ W^T  (+ rotary + rms-norm for q,k)
//   q,k out: bf16 [B][H][T][HD]      v out: bf16 [B][H][HD][T] (transposed)
// grid: (BT/64, NH, 3)  block: 128.  Each block: 64 rows x one 64-wide head.
// ---------------------------------------------------------------------------
__global__ __launch_bounds__(128) void qkv_rope_rms_kernel(
    const float* __restrict__ x, const float* __restrict__ cosb,
    const float* __restrict__ sinb, const float* __restrict__ wq,
    const float* __restrict__ wk, const float* __restrict__ wv,
    __bf16* __restrict__ qb, __bf16* __restrict__ kbuf, __bf16* __restrict__ vT)
{
  __shared__ alignas(32) __bf16 xs[64][40];      // 64 rows x 32 K (padded)
  __shared__ alignas(32) __bf16 wt[64][40];      // 64 cols x 32 K (padded)
  __shared__ alignas(32) float  res[64][HD + 1]; // result tile
  __shared__ float scale[64];

  const int tid  = threadIdx.x;
  const int wave = tid >> 5;
  const int lane = tid & 31;
  const int m    = lane & 15;
  const int half = lane >> 4;
  const int row0 = blockIdx.x << 6;   // global row in [0,BT), 64-aligned
  const int h    = blockIdx.y;
  const int which = blockIdx.z;       // 0=q 1=k 2=v
  const float* w = (which == 0) ? wq : (which == 1) ? wk : wv;

  v8f acc[4] = {};
  for (int k0 = 0; k0 < DIM; k0 += 32) {
    if (k0 + 32 < DIM) {  // stream-ahead hint -> global_prefetch_b8
      __builtin_prefetch(&x[(size_t)(row0 + (tid >> 1)) * DIM + k0 + 32], 0, 1);
      __builtin_prefetch(&w[(size_t)(h * HD + (tid >> 1)) * DIM + k0 + 32], 0, 1);
    }
    // stage x/w as 8-wide chunks: b128 loads + packed bf16 converts + b128 store
    for (int ch = tid; ch < 256; ch += 128) {
      int r = ch >> 2, c8 = (ch & 3) << 3;
      const float* src = &x[(size_t)(row0 + r) * DIM + k0 + c8];
      v8bf d;
#pragma unroll
      for (int j = 0; j < 8; ++j) d[j] = (__bf16)src[j];
      *(v8bf*)&xs[r][c8] = d;
    }
    for (int ch = tid; ch < 256; ch += 128) {
      int n = ch >> 2, c8 = (ch & 3) << 3;
      const float* src = &w[(size_t)(h * HD + n) * DIM + k0 + c8];
      v8bf d;
#pragma unroll
      for (int j = 0; j < 8; ++j) d[j] = (__bf16)src[j];
      *(v8bf*)&wt[n][c8] = d;
    }
    __syncthreads();
    v16bf af;
#pragma unroll
    for (int j = 0; j < 16; ++j) af[j] = xs[(wave << 4) + m][(half << 4) + j];
#pragma unroll
    for (int nt = 0; nt < 4; ++nt) {
      v16bf bf_;
#pragma unroll
      for (int j = 0; j < 16; ++j) bf_[j] = wt[(nt << 4) + m][(half << 4) + j];
      acc[nt] = wmma_bf16(af, bf_, acc[nt]);
    }
    __syncthreads();
  }

  // C layout: vgpr v -> row v + 8*half, col = lane&15 (within each 16-wide tile)
#pragma unroll
  for (int nt = 0; nt < 4; ++nt)
#pragma unroll
    for (int v = 0; v < 8; ++v)
      res[(wave << 4) + v + (half << 3)][(nt << 4) + m] = acc[nt][v];
  __syncthreads();

  const int b  = row0 >> 12;          // row0 / SEQ (64-row tile never crosses b)
  const int bh = b * NH + h;

  if (which < 2) {
    // rotary: pair c with c+32 within the 64-wide head
    for (int p = tid; p < 64 * 32; p += 128) {
      int r = p >> 5, c = p & 31;
      int t = (row0 + r) & (SEQ - 1);
      float cv = cosb[t * 32 + c], sv = sinb[t * 32 + c];
      float x1 = res[r][c], x2 = res[r][c + 32];
      res[r][c]      =  x1 * cv + x2 * sv;
      res[r][c + 32] = -x1 * sv + x2 * cv;
    }
    __syncthreads();
    if (tid < 64) {
      float ms = 0.f;
#pragma unroll
      for (int c = 0; c < HD; ++c) { float v = res[tid][c]; ms += v * v; }
      scale[tid] = rsqrtf(ms * (1.0f / HD) + EPSF);
    }
    __syncthreads();
    __bf16* outb = (which == 0) ? qb : kbuf;
    for (int e = tid; e < 64 * HD; e += 128) {
      int r = e >> 6, c = e & 63;
      int t = (row0 + r) & (SEQ - 1);
      outb[((size_t)bh * SEQ + t) * HD + c] = (__bf16)(res[r][c] * scale[r]);
    }
  } else {
    // V transposed: [B][H][HD][T]; adjacent lanes -> adjacent t (coalesced)
    for (int e = tid; e < 64 * HD; e += 128) {
      int c = e >> 6, r = e & 63;
      int t = (row0 + r) & (SEQ - 1);
      vT[((size_t)bh * HD + c) * SEQ + t] = (__bf16)res[r][c];
    }
  }
}

// ---------------------------------------------------------------------------
// Kernel 2: flash attention. 4 waves/block share async-LDS-staged K/V tiles.
// Each wave owns a 16-query tile; 32 keys/iter: 4 WMMA (S=QK^T) + 4 WMMA (PV).
// grid: (SEQ/64, NH, BATCH), block: 128
// ---------------------------------------------------------------------------
__global__ __launch_bounds__(128) void flash_attn_kernel(
    const __bf16* __restrict__ qb, const __bf16* __restrict__ kbuf,
    const __bf16* __restrict__ vT, float* __restrict__ y)
{
  __shared__ alignas(32) __bf16 kt[32][64];   // 32 keys x 64 hd
  __shared__ alignas(32) __bf16 vt[64][32];   // 64 hd x 32 keys (transposed)
  __shared__ float sp[4][16][33];             // per-wave S/P tile
  __shared__ float pmax[4][2][16], psum[4][2][16];
  __shared__ float mrow[4][16], lrow[4][16], arow[4][16];

  const int tid  = threadIdx.x;
  const int wave = tid >> 5;
  const int lane = tid & 31;
  const int m    = lane & 15;
  const int half = lane >> 4;
  const int t0   = (blockIdx.x << 6) + (wave << 4);
  const int h    = blockIdx.y;
  const int b    = blockIdx.z;
  const int bh   = b * NH + h;

  // Q fragments for this wave's 16 rows (registers, [B][H][T][HD] layout)
  const size_t qbase = ((size_t)bh * SEQ + t0 + m) * HD;
  v16bf qf[2];
#pragma unroll
  for (int c = 0; c < 2; ++c) {
    const __bf16* p = qb + qbase + c * 32 + (half << 4);
    v8bf lo = *(const v8bf*)(p);
    v8bf hi = *(const v8bf*)(p + 8);
#pragma unroll
    for (int j = 0; j < 8; ++j) { qf[c][j] = lo[j]; qf[c][j + 8] = hi[j]; }
  }

  v8f o[4] = {};
  if (lane < 16) { mrow[wave][m] = -3.0e38f; lrow[wave][m] = 0.f; }
  __syncthreads();

  const int kr = tid >> 2, kc = (tid & 3) << 4;   // K staging coords
  const int vr = tid >> 1, vc = (tid & 1) << 4;   // V staging coords

  for (int kb0 = 0; kb0 < SEQ; kb0 += 32) {
    // --- async global->LDS staging of K (32x64) and V^T (64x32), 32B/lane
    async_load_b128(&kt[kr][kc],
                    &kbuf[((size_t)bh * SEQ + kb0 + kr) * HD + kc]);
    async_load_b128(&vt[vr][vc],
                    &vT[((size_t)bh * HD + vr) * SEQ + kb0 + vc]);
    if (kb0 + 32 < SEQ)
      __builtin_prefetch(&kbuf[((size_t)bh * SEQ + kb0 + 32 + kr) * HD], 0, 1);
    wait_async0();
    __syncthreads();

    // --- S = Q K^T (two 16-key n-tiles, K-dim = 64 head dims in 2 chunks)
    v8f s0 = {}, s1 = {};
#pragma unroll
    for (int c = 0; c < 2; ++c) {
      v16bf kf = *(const v16bf*)&kt[m][c * 32 + (half << 4)];
      s0 = wmma_bf16(qf[c], kf, s0);
    }
#pragma unroll
    for (int c = 0; c < 2; ++c) {
      v16bf kf = *(const v16bf*)&kt[16 + m][c * 32 + (half << 4)];
      s1 = wmma_bf16(qf[c], kf, s1);
    }

    // --- spill scaled scores (1/sqrt(64)) to this wave's LDS tile
#pragma unroll
    for (int v = 0; v < 8; ++v) {
      sp[wave][v + (half << 3)][m]      = s0[v] * 0.125f;
      sp[wave][v + (half << 3)][16 + m] = s1[v] * 0.125f;
    }

    // --- online softmax: lane owns (row m, column half); all 32 lanes active
    float pm = -3.0e38f;
#pragma unroll
    for (int j = 0; j < 16; ++j) pm = fmaxf(pm, sp[wave][m][(half << 4) + j]);
    pmax[wave][half][m] = pm;
    float mp = mrow[wave][m];
    float mc = fmaxf(mp, fmaxf(pmax[wave][0][m], pmax[wave][1][m]));
    float al = __expf(mp - mc);
    float ps = 0.f;
#pragma unroll
    for (int j = 0; j < 16; ++j) {
      float p = __expf(sp[wave][m][(half << 4) + j] - mc);
      sp[wave][m][(half << 4) + j] = p;
      ps += p;
    }
    psum[wave][half][m] = ps;
    if (lane < 16) {
      lrow[wave][m] = lrow[wave][m] * al + psum[wave][0][m] + psum[wave][1][m];
      mrow[wave][m] = mc;
      arow[wave][m] = al;
    }

    // --- rescale running O by alpha(row)
#pragma unroll
    for (int v = 0; v < 8; ++v) {
      float a2 = arow[wave][v + (half << 3)];
      o[0][v] *= a2; o[1][v] *= a2; o[2][v] *= a2; o[3][v] *= a2;
    }

    // --- P as bf16 A-fragment (K = key index 0..31)
    v16bf pf;
#pragma unroll
    for (int j = 0; j < 16; ++j) pf[j] = (__bf16)sp[wave][m][(half << 4) + j];

    // --- O += P @ V : B-frag col n = head-dim, elem j = key half*16+j
#pragma unroll
    for (int nt = 0; nt < 4; ++nt) {
      v16bf vf = *(const v16bf*)&vt[(nt << 4) + m][half << 4];
      o[nt] = wmma_bf16(pf, vf, o[nt]);
    }
    __syncthreads();  // protect kt/vt for next iteration
  }

  // normalize and write fp32 y[B,T,DIM]
#pragma unroll
  for (int v = 0; v < 8; ++v) {
    int r = v + (half << 3);
    float inv = 1.0f / lrow[wave][r];
    size_t oidx = ((size_t)(b * SEQ) + t0 + r) * DIM + h * HD + m;
    y[oidx + 0]  = o[0][v] * inv;
    y[oidx + 16] = o[1][v] * inv;
    y[oidx + 32] = o[2][v] * inv;
    y[oidx + 48] = o[3][v] * inv;
  }
}

// ---------------------------------------------------------------------------
// Kernel 3: out = y @ Wo^T.  64x64 per block, 4 waves, 4 WMMA/wave/K-step.
// grid: (BT/64, DIM/64), block 128
// ---------------------------------------------------------------------------
__global__ __launch_bounds__(128) void out_proj_kernel(
    const float* __restrict__ y, const float* __restrict__ wo,
    float* __restrict__ out)
{
  __shared__ alignas(32) __bf16 xs[64][40];
  __shared__ alignas(32) __bf16 wt[64][40];
  const int tid  = threadIdx.x;
  const int wave = tid >> 5;
  const int lane = tid & 31;
  const int m    = lane & 15;
  const int half = lane >> 4;
  const int row0 = blockIdx.x << 6;
  const int n0   = blockIdx.y << 6;

  v8f acc[4] = {};
  for (int k0 = 0; k0 < DIM; k0 += 32) {
    if (k0 + 32 < DIM) {
      __builtin_prefetch(&y[(size_t)(row0 + (tid >> 1)) * DIM + k0 + 32], 0, 1);
      __builtin_prefetch(&wo[(size_t)(n0 + (tid >> 1)) * DIM + k0 + 32], 0, 1);
    }
    for (int ch = tid; ch < 256; ch += 128) {
      int r = ch >> 2, c8 = (ch & 3) << 3;
      const float* src = &y[(size_t)(row0 + r) * DIM + k0 + c8];
      v8bf d;
#pragma unroll
      for (int j = 0; j < 8; ++j) d[j] = (__bf16)src[j];
      *(v8bf*)&xs[r][c8] = d;
    }
    for (int ch = tid; ch < 256; ch += 128) {
      int n = ch >> 2, c8 = (ch & 3) << 3;
      const float* src = &wo[(size_t)(n0 + n) * DIM + k0 + c8];
      v8bf d;
#pragma unroll
      for (int j = 0; j < 8; ++j) d[j] = (__bf16)src[j];
      *(v8bf*)&wt[n][c8] = d;
    }
    __syncthreads();
    v16bf af;
#pragma unroll
    for (int j = 0; j < 16; ++j) af[j] = xs[(wave << 4) + m][(half << 4) + j];
#pragma unroll
    for (int nt = 0; nt < 4; ++nt) {
      v16bf bf_;
#pragma unroll
      for (int j = 0; j < 16; ++j) bf_[j] = wt[(nt << 4) + m][(half << 4) + j];
      acc[nt] = wmma_bf16(af, bf_, acc[nt]);
    }
    __syncthreads();
  }
#pragma unroll
  for (int nt = 0; nt < 4; ++nt)
#pragma unroll
    for (int v = 0; v < 8; ++v)
      out[(size_t)(row0 + (wave << 4) + v + (half << 3)) * DIM +
          n0 + (nt << 4) + m] = acc[nt][v];
}

// ---------------------------------------------------------------------------
extern "C" void kernel_launch(void* const* d_in, const int* in_sizes, int n_in,
                              void* d_out, int out_size, void* d_ws, size_t ws_size,
                              hipStream_t stream) {
  const float* x    = (const float*)d_in[0];
  const float* cosb = (const float*)d_in[1];
  const float* sinb = (const float*)d_in[2];
  const float* wq   = (const float*)d_in[3];
  const float* wk   = (const float*)d_in[4];
  const float* wv   = (const float*)d_in[5];
  const float* wo   = (const float*)d_in[6];
  float* out        = (float*)d_out;

  // workspace: q,k [B][H][T][HD] bf16, vT [B][H][HD][T] bf16, y [B,T,DIM] f32
  char* ws = (char*)d_ws;
  const size_t qkv_bytes = (size_t)BT * DIM * sizeof(__bf16);
  __bf16* qb = (__bf16*)(ws);
  __bf16* kb = (__bf16*)(ws + qkv_bytes);
  __bf16* vT = (__bf16*)(ws + 2 * qkv_bytes);
  float*  y  = (float*)(ws + 3 * qkv_bytes);

  qkv_rope_rms_kernel<<<dim3(BT / 64, NH, 3), 128, 0, stream>>>(
      x, cosb, sinb, wq, wk, wv, qb, kb, vT);
  flash_attn_kernel<<<dim3(SEQ / 64, NH, BATCH), 128, 0, stream>>>(qb, kb, vT, y);
  out_proj_kernel<<<dim3(BT / 64, DIM / 64), 128, 0, stream>>>(y, wo, out);
}
